// LlamaAttention_50483045597484
// MI455X (gfx1250) — matrix operand order repaired and checked by reading, not compiled, not run on hardware
//
#include <hip/hip_runtime.h>
#include <hip/hip_bf16.h>
#include <stdint.h>

// ---- problem shape ----
constexpr int SEQ  = 2048;
constexpr int HID  = 4096;
constexpr int NHQ  = 32;
constexpr int NKV  = 8;
constexpr int HD   = 128;
constexpr int QSZ  = NHQ * HD;          // 4096
constexpr int KVSZ = NKV * HD;          // 1024
constexpr int QKVN = QSZ + 2 * KVSZ;    // 6144

typedef __attribute__((ext_vector_type(16))) __bf16        v16bf;
typedef __attribute__((ext_vector_type(8)))  float         v8f;
typedef __attribute__((ext_vector_type(16))) unsigned short v16u;
typedef __attribute__((ext_vector_type(8)))  unsigned short v8u;

__device__ __forceinline__ unsigned short f2bf(float f) {
  unsigned int u = __float_as_uint(f);
  u += 0x7fffu + ((u >> 16) & 1u);      // round-to-nearest-even
  return (unsigned short)(u >> 16);
}
__device__ __forceinline__ v16bf as_bf(v16u v) {
  return __builtin_bit_cast(v16bf, v);
}

// CDNA5 async global->LDS copy (16B per lane), tracked by ASYNCcnt.
__device__ __forceinline__ void async_g2l_b128(unsigned lds_off, const void* gptr) {
  asm volatile("global_load_async_to_lds_b128 %0, %1, off"
               :: "v"(lds_off), "v"((unsigned long long)(uintptr_t)gptr)
               : "memory");
}
__device__ __forceinline__ void wait_async0() {
  asm volatile("s_wait_asynccnt 0x0" ::: "memory");
}
__device__ __forceinline__ unsigned lds_off_of(const void* p) {
  // AS(3)->flat addrspacecast keeps the LDS byte offset in the low 32 bits
  return (unsigned)(uintptr_t)p;
}

// ---------------- elementwise f32 -> bf16 (activations, row-major kept) ----------------
__global__ void cvt_bf16(const float* __restrict__ src,
                         unsigned short* __restrict__ dst, long n) {
  long i = (long)blockIdx.x * blockDim.x + threadIdx.x;
  long stride = (long)gridDim.x * blockDim.x;
  for (; i < n; i += stride) dst[i] = f2bf(src[i]);
}

// ---------------- f32 -> bf16 with transpose: dst[c*R + r] = bf16(src[r*C + c]) ----------
// 64x64 tiles through LDS so both global streams stay coalesced.
__global__ void cvt_bf16_t(const float* __restrict__ src,
                           unsigned short* __restrict__ dst, int R, int C) {
  __shared__ float tile[64][65];              // +1 pad: conflict-free column reads
  const int tid = threadIdx.x;                // 256 threads
  const int tx = tid & 63;
  const int ty = tid >> 6;                    // 0..3
  const int r0 = blockIdx.y * 64;
  const int c0 = blockIdx.x * 64;
#pragma unroll
  for (int i = 0; i < 16; ++i) {
    int row = ty + i * 4;
    tile[row][tx] = src[(size_t)(r0 + row) * C + c0 + tx];
  }
  __syncthreads();
#pragma unroll
  for (int i = 0; i < 16; ++i) {
    int row = ty + i * 4;                     // output row = source column c0+row
    dst[(size_t)(c0 + row) * R + r0 + tx] = f2bf(tile[tx][row]);
  }
}

// ---------------- bf16 WMMA GEMM: C[M,N] = A[M,K] @ Bt[N,K]^T ----------------
// Bt is N-major (pre-transposed weights), so BOTH fragment types are two runs of 8
// K-contiguous bf16 per lane -> pure global_load_b128 streams, no LDS needed.
// Block = 4 waves, 64(M) x 128(N); each wave: 4 M-subtiles x 2 N-subtiles,
// 8 WMMAs per 32-K step fed by 12 b128 loads.
__global__ void gemm_bf16_bt(const unsigned short* __restrict__ A,
                             const unsigned short* __restrict__ Bt,
                             float* __restrict__ C, int M, int N, int K) {
  const int lane = threadIdx.x & 31;
  const int wave = threadIdx.x >> 5;
  const int m  = lane & 15;
  const int hi = lane >> 4;                   // half-wave select
  const int row0 = blockIdx.y * 64;
  const int colw = blockIdx.x * 128 + wave * 32;

  v8f acc[4][2];
  v8f vzero = {};
#pragma unroll
  for (int mt = 0; mt < 4; ++mt)
#pragma unroll
    for (int t = 0; t < 2; ++t) acc[mt][t] = vzero;

  for (int k0 = 0; k0 < K; k0 += 32) {
    // A-frags: lane = row; elems 0..7 -> K=k0+hi*8+j, elems 8..15 -> +16
    v16bf af[4];
#pragma unroll
    for (int mt = 0; mt < 4; ++mt) {
      const unsigned short* ar = A + (size_t)(row0 + mt * 16 + m) * K + k0 + hi * 8;
      v8u lo = *(const v8u*)(ar);
      v8u h8 = *(const v8u*)(ar + 16);
      v16u au;
#pragma unroll
      for (int j = 0; j < 8; ++j) { au[j] = lo[j]; au[j + 8] = h8[j]; }
      af[mt] = as_bf(au);
    }
    // B-frags (Bt, N-major): lane = col; elems 0..7 -> K=k0+hi*8+j, elems 8..15 -> +16
#pragma unroll
    for (int t = 0; t < 2; ++t) {
      const unsigned short* br = Bt + (size_t)(colw + t * 16 + m) * K + k0 + hi * 8;
      v8u b0 = *(const v8u*)(br);
      v8u b1 = *(const v8u*)(br + 16);
      v16u bu;
#pragma unroll
      for (int j = 0; j < 8; ++j) { bu[j] = b0[j]; bu[j + 8] = b1[j]; }
      v16bf bf = as_bf(bu);
#pragma unroll
      for (int mt = 0; mt < 4; ++mt)
        acc[mt][t] = __builtin_amdgcn_wmma_f32_16x16x32_bf16(false, af[mt], false, bf,
                                                             (short)0, acc[mt][t], false, false);
    }
  }
  // D layout: lane = col n, rows r + hi*8 across the 8 accumulator regs
#pragma unroll
  for (int mt = 0; mt < 4; ++mt)
#pragma unroll
    for (int t = 0; t < 2; ++t)
#pragma unroll
      for (int r = 0; r < 8; ++r)
        C[(size_t)(row0 + mt * 16 + r + hi * 8) * N + colw + t * 16 + m] = acc[mt][t][r];
}

// ---------------- RoPE (neox) + split qkv into bf16 Q/K/V ----------------
__global__ void rope_qkv(const int* __restrict__ pos,
                         const float* __restrict__ qkv,
                         unsigned short* __restrict__ qb,
                         unsigned short* __restrict__ kbuf,
                         unsigned short* __restrict__ vbuf) {
  const int s = blockIdx.x;
  const float p = (float)pos[s];
  const float* row = qkv + (size_t)s * QKVN;
  for (int t = threadIdx.x; t < 3584; t += blockDim.x) {
    if (t < 2560) {
      const int head = t >> 6;          // 0..39 (q heads 0..31, k heads 32..39)
      const int i = t & 63;
      const float* hb = row + head * HD;   // q then k contiguous in qkv row
      float x1 = hb[i], x2 = hb[i + 64];
      float freq = __powf(10000.0f, -(float)i * (1.0f / 64.0f));
      float ang = p * freq;
      float sn, cs;
      __sincosf(ang, &sn, &cs);
      float o1 = x1 * cs - x2 * sn;
      float o2 = x2 * cs + x1 * sn;
      if (head < NHQ) {
        unsigned short* q = qb + (size_t)s * QSZ + head * HD;
        q[i] = f2bf(o1); q[i + 64] = f2bf(o2);
      } else {
        unsigned short* k = kbuf + (size_t)s * KVSZ + (head - NHQ) * HD;
        k[i] = f2bf(o1); k[i + 64] = f2bf(o2);
      }
    } else {
      const int u = t - 2560;
      vbuf[(size_t)s * KVSZ + u] = f2bf(row[QSZ + KVSZ + u]);
    }
  }
}

// ---------------- flash attention, one wave per (head, 16-row query block) ----------------
// Block = 4 waves sharing one head (and kv head); V blocks staged via async copies.
__global__ void attn_fwd(const unsigned short* __restrict__ qb,
                         const unsigned short* __restrict__ kbuf,
                         const unsigned short* __restrict__ vbuf,
                         unsigned short* __restrict__ ob) {
  __shared__ float pbuf[4][16 * 32];          // per-wave P staging (8KB)
  __shared__ unsigned short vs[32 * 128];     // V block, key-major rows (8KB)
  const int tid  = threadIdx.x;
  const int lane = tid & 31;
  const int wave = tid >> 5;
  const int h    = blockIdx.x;                // 0..31
  const int qblk = blockIdx.y * 4 + wave;     // 0..127
  const int kvh  = h >> 2;                    // GQA: 4 q heads per kv head
  const int m  = lane & 15;
  const int hi = lane >> 4;
  const float scale = 0.08838834764831845f;   // 1/sqrt(128)

  // Q fragments: 16 rows x 128 dims as four K=32 A-frags (contiguous 16B loads)
  v16bf qf[4];
  {
    const unsigned short* qrow = qb + (size_t)(qblk * 16 + m) * QSZ + h * HD;
#pragma unroll
    for (int c = 0; c < 4; ++c) {
      v8u lo = *(const v8u*)(qrow + c * 32 + hi * 8);
      v8u hi8 = *(const v8u*)(qrow + c * 32 + hi * 8 + 16);
      v16u u;
#pragma unroll
      for (int j = 0; j < 8; ++j) { u[j] = lo[j]; u[j + 8] = hi8[j]; }
      qf[c] = as_bf(u);
    }
  }

  v8f o[8];
  v8f vzero = {};
#pragma unroll
  for (int t = 0; t < 8; ++t) o[t] = vzero;
  float mrow[8], lrow[8];
#pragma unroll
  for (int r = 0; r < 8; ++r) { mrow[r] = -1e30f; lrow[r] = 0.0f; }

  const int qend = qblk * 16 + 15;                    // this wave's causal bound
  const int qmax = (blockIdx.y * 4 + 3) * 16 + 15;    // block-uniform loop bound
  float* pl = pbuf[wave];

  for (int kb0 = 0; kb0 <= qmax; kb0 += 32) {
    // ---- cooperative async stage of V[kb0:kb0+32, head dims 0:128) ----
    __syncthreads();                          // WAR on vs
#pragma unroll
    for (int i = 0; i < 4; ++i) {
      int seg = tid + i * 128;                // 512 x 16B segments
      int key = seg >> 4;                     // 0..31
      int d8  = (seg & 15) * 8;               // 0..120
      async_g2l_b128(lds_off_of(&vs[key * 128 + d8]),
                     vbuf + (size_t)(kb0 + key) * KVSZ + kvh * HD + d8);
    }
    wait_async0();
    __syncthreads();                          // vs visible to all waves

    if (kb0 <= qend) {                        // wave-uniform causal skip
      // ---- scores: S(16x32) = Q(16x128) @ K^T, two 16x16 D tiles ----
      v8f sc0 = vzero, sc1 = vzero;
#pragma unroll
      for (int c = 0; c < 4; ++c) {
        const unsigned short* k0p = kbuf + (size_t)(kb0 + m) * KVSZ + kvh * HD + c * 32 + hi * 8;
        const unsigned short* k1p = kbuf + (size_t)(kb0 + 16 + m) * KVSZ + kvh * HD + c * 32 + hi * 8;
        v8u a0  = *(const v8u*)(k0p);
        v8u a0b = *(const v8u*)(k0p + 16);
        v8u a1  = *(const v8u*)(k1p);
        v8u a1b = *(const v8u*)(k1p + 16);
        v16u b0, b1;
#pragma unroll
        for (int j = 0; j < 8; ++j) { b0[j] = a0[j]; b0[j + 8] = a0b[j];
                                      b1[j] = a1[j]; b1[j + 8] = a1b[j]; }
        sc0 = __builtin_amdgcn_wmma_f32_16x16x32_bf16(false, qf[c], false, as_bf(b0),
                                                      (short)0, sc0, false, false);
        sc1 = __builtin_amdgcn_wmma_f32_16x16x32_bf16(false, qf[c], false, as_bf(b1),
                                                      (short)0, sc1, false, false);
      }

      // ---- online softmax over the 16x32 block ----
#pragma unroll
      for (int r = 0; r < 8; ++r) {
        const int qrow = qblk * 16 + r + hi * 8;
        const int key0 = kb0 + m;
        const int key1 = kb0 + 16 + m;
        float s0 = (key0 <= qrow) ? sc0[r] * scale : -1e30f;
        float s1 = (key1 <= qrow) ? sc1[r] * scale : -1e30f;
        float mr = fmaxf(s0, s1);
#pragma unroll
        for (int off = 8; off >= 1; off >>= 1)
          mr = fmaxf(mr, __shfl_xor(mr, off, 16));
        float newm = fmaxf(mrow[r], mr);
        float corr = __expf(mrow[r] - newm);
        float p0 = __expf(s0 - newm);
        float p1 = __expf(s1 - newm);
        float rs = p0 + p1;
#pragma unroll
        for (int off = 8; off >= 1; off >>= 1)
          rs += __shfl_xor(rs, off, 16);
        lrow[r] = lrow[r] * corr + rs;
        mrow[r] = newm;
#pragma unroll
        for (int t = 0; t < 8; ++t) o[t][r] = o[t][r] * corr;
        pl[(r + hi * 8) * 32 + m]      = p0;   // P in D-layout
        pl[(r + hi * 8) * 32 + 16 + m] = p1;
      }
      asm volatile("s_wait_dscnt 0" ::: "memory");   // in-wave LDS transpose fence

      // ---- P A-frag (16x32): lane = row m; elems 0..7 -> k=hi*8+j, 8..15 -> k=16+hi*8+j
      v16u pu;
#pragma unroll
      for (int j = 0; j < 8; ++j) {
        pu[j]     = f2bf(pl[m * 32 + hi * 8 + j]);
        pu[j + 8] = f2bf(pl[m * 32 + 16 + hi * 8 + j]);
      }
      v16bf pf = as_bf(pu);

      // ---- O(16x128) += P(16x32) @ V(32x128): V B-frags gathered from LDS ----
#pragma unroll
      for (int t = 0; t < 8; ++t) {
        v16u vu;
#pragma unroll
        for (int j = 0; j < 16; ++j)
          vu[j] = vs[(hi * 8 + (j & 7) + (j >> 3) * 16) * 128 + t * 16 + m];
        o[t] = __builtin_amdgcn_wmma_f32_16x16x32_bf16(false, pf, false, as_bf(vu),
                                                       (short)0, o[t], false, false);
      }
    }
  }

  // ---- normalize and emit bf16 attention output ----
#pragma unroll
  for (int r = 0; r < 8; ++r) {
    float inv = 1.0f / lrow[r];
    size_t rowoff = (size_t)(qblk * 16 + r + hi * 8) * QSZ + h * HD;
#pragma unroll
    for (int t = 0; t < 8; ++t)
      ob[rowoff + t * 16 + m] = f2bf(o[t][r] * inv);
  }
}

// ---------------- launch ----------------
extern "C" void kernel_launch(void* const* d_in, const int* in_sizes, int n_in,
                              void* d_out, int out_size, void* d_ws, size_t ws_size,
                              hipStream_t stream) {
  const int*   positions = (const int*)d_in[0];
  const float* hidden    = (const float*)d_in[1];
  const float* w_qkv     = (const float*)d_in[2];
  const float* w_o       = (const float*)d_in[3];
  float*       out       = (float*)d_out;

  char* ws = (char*)d_ws;
  size_t off = 0;
  auto alloc = [&](size_t bytes) -> char* {
    char* p = ws + off;
    off += (bytes + 255) & ~(size_t)255;
    return p;
  };
  unsigned short* hb   = (unsigned short*)alloc((size_t)SEQ * HID * 2);   // A, row-major
  unsigned short* wqt  = (unsigned short*)alloc((size_t)HID * QKVN * 2);  // w_qkv^T, N-major
  unsigned short* wot  = (unsigned short*)alloc((size_t)QSZ * HID * 2);   // w_o^T,  N-major
  float*          qkv  = (float*)alloc((size_t)SEQ * QKVN * 4);
  unsigned short* qb   = (unsigned short*)alloc((size_t)SEQ * QSZ * 2);
  unsigned short* kbuf = (unsigned short*)alloc((size_t)SEQ * KVSZ * 2);
  unsigned short* vbuf = (unsigned short*)alloc((size_t)SEQ * KVSZ * 2);
  unsigned short* ab   = (unsigned short*)alloc((size_t)SEQ * QSZ * 2);

  cvt_bf16<<<2048, 256, 0, stream>>>(hidden, hb, (long)SEQ * HID);
  cvt_bf16_t<<<dim3(QKVN / 64, HID / 64), 256, 0, stream>>>(w_qkv, wqt, HID, QKVN);
  cvt_bf16_t<<<dim3(HID / 64, QSZ / 64), 256, 0, stream>>>(w_o, wot, QSZ, HID);

  gemm_bf16_bt<<<dim3(QKVN / 128, SEQ / 64), 128, 0, stream>>>(hb, wqt, qkv, SEQ, QKVN, HID);
  rope_qkv<<<SEQ, 256, 0, stream>>>(positions, qkv, qb, kbuf, vbuf);
  attn_fwd<<<dim3(NHQ, SEQ / 16 / 4), 128, 0, stream>>>(qb, kbuf, vbuf, ab);
  gemm_bf16_bt<<<dim3(HID / 128, SEQ / 64), 128, 0, stream>>>(ab, wot, out, SEQ, HID, QSZ);
}
